// AttentionFlow_80805514707060
// MI455X (gfx1250) — matrix operand order repaired and checked by reading, not compiled, not run on hardware
//
#include <hip/hip_runtime.h>

// ---------------- constants (fixed by the reference) ----------------
#define N_NODES  100000
#define N_QUERIES 128
#define DIM      128
#define E_EDGES  131072
#define TK_CAP   2048

// ---------------- WMMA types / helpers ----------------
typedef __attribute__((ext_vector_type(16))) __bf16 v16bf;
typedef __attribute__((ext_vector_type(8)))  float  v8f;

union ABFrag {
  v16bf v;
  uint4 q[2];
  unsigned short u[16];
};

__device__ __forceinline__ v8f wmma_bf16(const ABFrag& a, const ABFrag& b, v8f c) {
  return __builtin_amdgcn_wmma_f32_16x16x32_bf16(false, a.v, false, b.v,
                                                 (short)0, c, false, false);
}

__device__ __forceinline__ unsigned short f2bf(float f) {
  unsigned u = __float_as_uint(f);
  unsigned r = ((u >> 16) & 1u) + 0x7FFFu;   // round to nearest even
  return (unsigned short)((u + r) >> 16);
}

__device__ __forceinline__ unsigned long long pack4bf(float4 v) {
  return (unsigned long long)f2bf(v.x)
       | ((unsigned long long)f2bf(v.y) << 16)
       | ((unsigned long long)f2bf(v.z) << 32)
       | ((unsigned long long)f2bf(v.w) << 48);
}

__device__ __forceinline__ float leaky(float x) { return (x > 0.f) ? x : 0.01f * x; }

__device__ __forceinline__ void atomicMaxF(float* addr, float v) {
  if (v >= 0.f) atomicMax((int*)addr, __float_as_int(v));
  else          atomicMin((unsigned int*)addr, __float_as_uint(v));
}

// ---------------- weight packing: W[N,K] f32 -> bf16 B-fragment blocks ----
// block b = j*(K/32)+kk : 32 lanes x 16 bf16; lane gives col n=j*16+(lane&15),
// k = kk*32 + (lane>>4)*8 + t (t<8), kk*32 + 16 + (lane>>4)*8 + (t-8) (t>=8)
__global__ void pack_w_kernel(const float* __restrict__ W,
                              unsigned short* __restrict__ P,
                              int Ndim, int Kdim) {
  int gid = blockIdx.x * blockDim.x + threadIdx.x;
  int nkk = Kdim >> 5;
  int total = (Ndim >> 4) * nkk * 32;
  if (gid >= total) return;
  int blk = gid >> 5, lane = gid & 31;
  int kk = blk % nkk, j = blk / nkk;
  int n  = j * 16 + (lane & 15);
  int kb = kk * 32 + (lane >> 4) * 8;
  unsigned short* dst = P + (size_t)blk * 512 + lane * 16;
  const float* src = W + (size_t)n * Kdim + kb;
#pragma unroll
  for (int t = 0; t < 16; ++t) dst[t] = f2bf(src[t + ((t >> 3) << 3)]);
}

// ---------------- edge kernel: 1 wave handles 32 edges (2 M-tiles) -------
// Each B fragment is loaded once and feeds two WMMAs (doubles FLOP/L2-byte).
// X layout (bf16 in LDS): [hi 0..127 | hj 128..255 | rel 256..383 | qs 384..511 | qr 512..639]
__global__ __launch_bounds__(32) void edge_kernel(
    const float* __restrict__ node_rep, const float* __restrict__ qse,
    const float* __restrict__ qre, const float* __restrict__ rel_emb,
    const float* __restrict__ b_left, const float* __restrict__ b_right,
    const float* __restrict__ b_center,
    const unsigned short* __restrict__ WLp, const unsigned short* __restrict__ WRp,
    const unsigned short* __restrict__ WCp,
    const int* __restrict__ query_idx, const int* __restrict__ idx_i,
    const int* __restrict__ idx_j, float* __restrict__ logits) {
  __shared__ alignas(16) unsigned short Xs[32][648];  // 640 + 8 pad
  __shared__ alignas(16) unsigned short Rs[32][264];  // 256 + 8 pad
  const int tile = blockIdx.x;
  const int lane = threadIdx.x;
  const int lo = lane & 15, half = lane >> 4;

  // ---- gather + bf16 convert into LDS (coalesced float4 per lane) ----
  for (int m = 0; m < 32; ++m) {
    const int e = tile * 32 + m;
    const int ii = idx_i[e], jj = idx_j[e], qi = query_idx[e];
    const float4 hi4 = ((const float4*)(node_rep + (size_t)ii * DIM))[lane];
    const float4 hj4 = ((const float4*)(node_rep + (size_t)jj * DIM))[lane];
    const float4 re4 = ((const float4*)(rel_emb + (size_t)e  * DIM))[lane];
    const float4 qs4 = ((const float4*)(qse + (size_t)qi * DIM))[lane];
    const float4 qr4 = ((const float4*)(qre + (size_t)qi * DIM))[lane];
    const int c0 = lane * 4;
    *(unsigned long long*)&Xs[m][c0]       = pack4bf(hi4);
    *(unsigned long long*)&Xs[m][128 + c0] = pack4bf(hj4);
    *(unsigned long long*)&Xs[m][256 + c0] = pack4bf(re4);
    *(unsigned long long*)&Xs[m][384 + c0] = pack4bf(qs4);
    *(unsigned long long*)&Xs[m][512 + c0] = pack4bf(qr4);
  }
  __syncthreads();

  // ---- step 1: R = leaky(rightX @ Wr^T + br), stored bf16 in Rs ----
#pragma unroll 1
  for (int j = 0; j < 16; ++j) {
    const float bb = b_right[j * 16 + lo];
    v8f c0 = {bb, bb, bb, bb, bb, bb, bb, bb};
    v8f c1 = c0;
#pragma unroll
    for (int kk = 0; kk < 16; ++kk) {
      const int cb = 128 + kk * 32;  // right K k maps to X col k+128
      ABFrag a0, a1, b;
      const uint4* wp = (const uint4*)(WRp + (size_t)(j * 16 + kk) * 512);
      b.q[0] = wp[lane * 2];
      b.q[1] = wp[lane * 2 + 1];
      a0.q[0] = *(const uint4*)&Xs[lo][cb + half * 8];
      a0.q[1] = *(const uint4*)&Xs[lo][cb + 16 + half * 8];
      a1.q[0] = *(const uint4*)&Xs[16 + lo][cb + half * 8];
      a1.q[1] = *(const uint4*)&Xs[16 + lo][cb + 16 + half * 8];
      c0 = wmma_bf16(a0, b, c0);
      c1 = wmma_bf16(a1, b, c1);
    }
#pragma unroll
    for (int v = 0; v < 8; ++v) {
      Rs[v + half * 8][j * 16 + lo]      = f2bf(leaky(c0[v]));
      Rs[16 + v + half * 8][j * 16 + lo] = f2bf(leaky(c1[v]));
    }
  }
  __syncthreads();

  // ---- step 2: l = leaky(leftX @ Wl^T + bl); r2 = R @ Wc^T + bc; logits += l.r2 ----
  float p0[8] = {0, 0, 0, 0, 0, 0, 0, 0};
  float p1[8] = {0, 0, 0, 0, 0, 0, 0, 0};
#pragma unroll 1
  for (int j = 0; j < 16; ++j) {
    const float bl = b_left[j * 16 + lo];
    v8f cl0 = {bl, bl, bl, bl, bl, bl, bl, bl};
    v8f cl1 = cl0;
#pragma unroll
    for (int kk = 0; kk < 16; ++kk) {
      const int cb = kk * 32 + ((kk >= 4) ? 128 : 0);  // left K: hi then rel/qs/qr
      ABFrag a0, a1, b;
      const uint4* wp = (const uint4*)(WLp + (size_t)(j * 16 + kk) * 512);
      b.q[0] = wp[lane * 2];
      b.q[1] = wp[lane * 2 + 1];
      a0.q[0] = *(const uint4*)&Xs[lo][cb + half * 8];
      a0.q[1] = *(const uint4*)&Xs[lo][cb + 16 + half * 8];
      a1.q[0] = *(const uint4*)&Xs[16 + lo][cb + half * 8];
      a1.q[1] = *(const uint4*)&Xs[16 + lo][cb + 16 + half * 8];
      cl0 = wmma_bf16(a0, b, cl0);
      cl1 = wmma_bf16(a1, b, cl1);
    }
    const float bc = b_center[j * 16 + lo];
    v8f cr0 = {bc, bc, bc, bc, bc, bc, bc, bc};
    v8f cr1 = cr0;
#pragma unroll
    for (int kk = 0; kk < 8; ++kk) {
      ABFrag a0, a1, b;
      const uint4* wp = (const uint4*)(WCp + (size_t)(j * 8 + kk) * 512);
      b.q[0] = wp[lane * 2];
      b.q[1] = wp[lane * 2 + 1];
      a0.q[0] = *(const uint4*)&Rs[lo][kk * 32 + half * 8];
      a0.q[1] = *(const uint4*)&Rs[lo][kk * 32 + 16 + half * 8];
      a1.q[0] = *(const uint4*)&Rs[16 + lo][kk * 32 + half * 8];
      a1.q[1] = *(const uint4*)&Rs[16 + lo][kk * 32 + 16 + half * 8];
      cr0 = wmma_bf16(a0, b, cr0);
      cr1 = wmma_bf16(a1, b, cr1);
    }
#pragma unroll
    for (int v = 0; v < 8; ++v) {
      p0[v] += leaky(cl0[v]) * cr0[v];
      p1[v] += leaky(cl1[v]) * cr1[v];
    }
  }
  // reduce over the 16-lane N dimension (stays within each half of wave32)
#pragma unroll
  for (int off = 1; off < 16; off <<= 1) {
#pragma unroll
    for (int v = 0; v < 8; ++v) {
      p0[v] += __shfl_xor(p0[v], off, 32);
      p1[v] += __shfl_xor(p1[v], off, 32);
    }
  }
  if (lo == 0) {
#pragma unroll
    for (int v = 0; v < 8; ++v) {
      logits[tile * 32 + v + half * 8]      = p0[v];
      logits[tile * 32 + 16 + v + half * 8] = p1[v];
    }
  }
}

// ---------------- segment softmax over idx_i ----------------
__global__ void seg_init_kernel(float* smax, float* sden, int n) {
  int i = blockIdx.x * blockDim.x + threadIdx.x;
  if (i < n) { smax[i] = __uint_as_float(0xFF800000u); sden[i] = 0.f; }
}

__global__ void smax_kernel(const float* __restrict__ logits,
                            const int* __restrict__ idx_i, float* smax) {
  int e = blockIdx.x * blockDim.x + threadIdx.x;
  if (e < E_EDGES) atomicMaxF(&smax[idx_i[e]], logits[e]);
}

__global__ void sexp_kernel(const float* __restrict__ logits,
                            const int* __restrict__ idx_i,
                            const float* __restrict__ smax,
                            float* __restrict__ attn, float* sden) {
  int e = blockIdx.x * blockDim.x + threadIdx.x;
  if (e < E_EDGES) {
    float ex = __expf(logits[e] - smax[idx_i[e]]);
    attn[e] = ex;
    atomicAdd(&sden[idx_i[e]], ex);
  }
}

__global__ void snorm_kernel(const int* __restrict__ idx_i,
                             const float* __restrict__ sden,
                             const float* __restrict__ visited,
                             float* __restrict__ attn, float* __restrict__ tsc) {
  int e = blockIdx.x * blockDim.x + threadIdx.x;
  if (e < E_EDGES) {
    int i = idx_i[e];
    float a = attn[e] / sden[i];
    attn[e] = a;
    tsc[e] = a * visited[i];
  }
}

// ---------------- per-query top-k (keep rank < max_edges) ----------------
__global__ __launch_bounds__(256) void topk_kernel(
    const float* __restrict__ tsc, const int* __restrict__ qidx,
    const int* __restrict__ maxe_p, float* __restrict__ keepw) {
  __shared__ unsigned long long keys[TK_CAP];
  __shared__ int cntS;
  const int q = blockIdx.x, tid = threadIdx.x;
  const int maxe = maxe_p[0];
  if (tid == 0) cntS = 0;
  __syncthreads();
  for (int e = tid; e < E_EDGES; e += 256) {
    if (qidx[e] == q) {
      int p = atomicAdd(&cntS, 1);
      if (p < TK_CAP) {
        unsigned u = __float_as_uint(tsc[e]);
        unsigned k = (u & 0x80000000u) ? ~u : (u | 0x80000000u);   // ascending order key
        keys[p] = ((unsigned long long)(~k) << 32) | (unsigned)e;  // sort asc = score desc, idx asc
      }
    }
  }
  __syncthreads();
  int cnt = cntS < TK_CAP ? cntS : TK_CAP;
  if (cnt <= maxe) {  // uniform branch: keep everything in this group
    for (int e = tid; e < E_EDGES; e += 256)
      if (qidx[e] == q) keepw[e] = 1.0f;
    return;
  }
  for (int p = cnt + tid; p < TK_CAP; p += 256) keys[p] = ~0ULL;
  __syncthreads();
  for (unsigned k = 2; k <= TK_CAP; k <<= 1) {
    for (unsigned j = k >> 1; j > 0; j >>= 1) {
      for (unsigned idx = tid; idx < TK_CAP; idx += 256) {
        unsigned ixj = idx ^ j;
        if (ixj > idx) {
          unsigned long long a = keys[idx], b = keys[ixj];
          bool asc = ((idx & k) == 0);
          if ((a > b) == asc) { keys[idx] = b; keys[ixj] = a; }
        }
      }
      __syncthreads();
    }
  }
  for (int p = tid; p < maxe; p += 256) {
    unsigned long long kk = keys[p];
    if (kk != ~0ULL) keepw[(unsigned)(kk & 0xFFFFFFFFu)] = 1.0f;
  }
}

// ---------------- sparse scatter: scores + attention-weighted messages ----
__global__ __launch_bounds__(128) void agg_kernel(
    const float* __restrict__ node_rep, const float* __restrict__ attn,
    const float* __restrict__ tsc, const float* __restrict__ keepw,
    const int* __restrict__ idx_i, const int* __restrict__ idx_j,
    float* __restrict__ score_out, float* __restrict__ rep_acc) {
  const int e = blockIdx.x, d = threadIdx.x;
  const float kw = keepw[e];
  if (kw == 0.f) return;
  const int i = idx_i[e], j = idx_j[e];
  const float a = attn[e] * kw;
  atomicAdd(&rep_acc[(size_t)j * DIM + d], a * node_rep[(size_t)i * DIM + d]);
  if (d == 0) atomicAdd(&score_out[j], kw * tsc[e]);
}

// ---------------- final: out = leaky((node_rep + agg) @ Wstep^T + b), in-place ----
__global__ __launch_bounds__(32) void step_kernel(
    const float* __restrict__ node_rep, const float* __restrict__ b_step,
    const unsigned short* __restrict__ WSp, float* __restrict__ rep_io) {
  __shared__ alignas(16) unsigned short As[32][136];  // 128 + 8 pad
  const int t = blockIdx.x;
  const int lane = threadIdx.x;
  const int lo = lane & 15, half = lane >> 4;
  const int r0 = t * 32;
  for (int m = 0; m < 32; ++m) {
    const int row = r0 + m;  // N_NODES is a multiple of 32
    float4 a4 = ((const float4*)(node_rep + (size_t)row * DIM))[lane];
    float4 g4 = ((const float4*)(rep_io + (size_t)row * DIM))[lane];
    a4.x += g4.x; a4.y += g4.y; a4.z += g4.z; a4.w += g4.w;
    *(unsigned long long*)&As[m][lane * 4] = pack4bf(a4);
  }
  __syncthreads();
#pragma unroll 1
  for (int j = 0; j < 8; ++j) {
    const float bs = b_step[j * 16 + lo];
    v8f c0 = {bs, bs, bs, bs, bs, bs, bs, bs};
    v8f c1 = c0;
#pragma unroll
    for (int kk = 0; kk < 4; ++kk) {
      ABFrag a0, a1, b;
      const uint4* wp = (const uint4*)(WSp + (size_t)(j * 4 + kk) * 512);
      b.q[0] = wp[lane * 2];
      b.q[1] = wp[lane * 2 + 1];
      a0.q[0] = *(const uint4*)&As[lo][kk * 32 + half * 8];
      a0.q[1] = *(const uint4*)&As[lo][kk * 32 + 16 + half * 8];
      a1.q[0] = *(const uint4*)&As[16 + lo][kk * 32 + half * 8];
      a1.q[1] = *(const uint4*)&As[16 + lo][kk * 32 + 16 + half * 8];
      c0 = wmma_bf16(a0, b, c0);
      c1 = wmma_bf16(a1, b, c1);
    }
#pragma unroll
    for (int v = 0; v < 8; ++v) {
      rep_io[(size_t)(r0 + v + half * 8) * DIM + j * 16 + lo]      = leaky(c0[v]);
      rep_io[(size_t)(r0 + 16 + v + half * 8) * DIM + j * 16 + lo] = leaky(c1[v]);
    }
  }
}

// ---------------- host side ----------------
static inline void* ws_alloc(void* base, size_t& off, size_t bytes) {
  off = (off + 255) & ~(size_t)255;
  void* p = (char*)base + off;
  off += bytes;
  return p;
}

extern "C" void kernel_launch(void* const* d_in, const int* in_sizes, int n_in,
                              void* d_out, int out_size, void* d_ws, size_t ws_size,
                              hipStream_t stream) {
  (void)in_sizes; (void)n_in; (void)out_size; (void)ws_size;
  const float* visited  = (const float*)d_in[0];
  const float* node_rep = (const float*)d_in[1];
  const float* qse      = (const float*)d_in[2];
  const float* qre      = (const float*)d_in[3];
  const float* rel_emb  = (const float*)d_in[4];
  const float* W_left   = (const float*)d_in[5];
  const float* b_left   = (const float*)d_in[6];
  const float* W_right  = (const float*)d_in[7];
  const float* b_right  = (const float*)d_in[8];
  const float* W_center = (const float*)d_in[9];
  const float* b_center = (const float*)d_in[10];
  const float* W_step   = (const float*)d_in[11];
  const float* b_step   = (const float*)d_in[12];
  const int* query_idx  = (const int*)d_in[13];
  const int* idx_i      = (const int*)d_in[14];
  const int* idx_j      = (const int*)d_in[15];
  const int* max_edges  = (const int*)d_in[16];

  float* out_score = (float*)d_out;            // [N_NODES]
  float* out_rep   = out_score + N_NODES;      // [N_NODES, 128] (also agg accumulator)

  size_t o = 0;
  unsigned short* WLp = (unsigned short*)ws_alloc(d_ws, o, 256 * 512 * 2);
  unsigned short* WRp = (unsigned short*)ws_alloc(d_ws, o, 256 * 512 * 2);
  unsigned short* WCp = (unsigned short*)ws_alloc(d_ws, o, 256 * 256 * 2);
  unsigned short* WSp = (unsigned short*)ws_alloc(d_ws, o, 128 * 128 * 2);
  float* logits = (float*)ws_alloc(d_ws, o, (size_t)E_EDGES * 4);
  float* attn   = (float*)ws_alloc(d_ws, o, (size_t)E_EDGES * 4);
  float* tsc    = (float*)ws_alloc(d_ws, o, (size_t)E_EDGES * 4);
  float* keepw  = (float*)ws_alloc(d_ws, o, (size_t)E_EDGES * 4);
  float* smax   = (float*)ws_alloc(d_ws, o, (size_t)N_NODES * 4);
  float* sden   = (float*)ws_alloc(d_ws, o, (size_t)N_NODES * 4);

  // 1) pack weights into WMMA B-fragment order (bf16)
  pack_w_kernel<<<(16 * 16 * 32 + 255) / 256, 256, 0, stream>>>(W_left,   WLp, 256, 512);
  pack_w_kernel<<<(16 * 16 * 32 + 255) / 256, 256, 0, stream>>>(W_right,  WRp, 256, 512);
  pack_w_kernel<<<(16 *  8 * 32 + 255) / 256, 256, 0, stream>>>(W_center, WCp, 256, 256);
  pack_w_kernel<<<( 8 *  4 * 32 + 255) / 256, 256, 0, stream>>>(W_step,   WSp, 128, 128);

  // 2) zero outputs (scores + agg accumulator) and keep flags
  hipMemsetAsync(d_out, 0, (size_t)(N_NODES + (size_t)N_NODES * DIM) * 4, stream);
  hipMemsetAsync(keepw, 0, (size_t)E_EDGES * 4, stream);
  seg_init_kernel<<<(N_NODES + 255) / 256, 256, 0, stream>>>(smax, sden, N_NODES);

  // 3) fused bilinear edge scorer (WMMA bf16, fp32 accumulate, 2 M-tiles/wave)
  edge_kernel<<<E_EDGES / 32, 32, 0, stream>>>(
      node_rep, qse, qre, rel_emb, b_left, b_right, b_center,
      WLp, WRp, WCp, query_idx, idx_i, idx_j, logits);

  // 4) segment softmax + target scores
  smax_kernel<<<E_EDGES / 256, 256, 0, stream>>>(logits, idx_i, smax);
  sexp_kernel<<<E_EDGES / 256, 256, 0, stream>>>(logits, idx_i, smax, attn, sden);
  snorm_kernel<<<E_EDGES / 256, 256, 0, stream>>>(idx_i, sden, visited, attn, tsc);

  // 5) per-query top-k pruning
  topk_kernel<<<N_QUERIES, 256, 0, stream>>>(tsc, query_idx, max_edges, keepw);

  // 6) pruned scatter of scores and attention-weighted messages
  agg_kernel<<<E_EDGES, 128, 0, stream>>>(node_rep, attn, tsc, keepw,
                                          idx_i, idx_j, out_score, out_rep);

  // 7) final linear-between-steps + LeakyReLU (in-place over out_rep, WMMA, 2 M-tiles/wave)
  step_kernel<<<N_NODES / 32, 32, 0, stream>>>(node_rep, b_step, WSp, out_rep);
}
